// GCN_62483184222290
// MI455X (gfx1250) — compile-verified, matrix operand order checked by reading, not run on hardware
//
#include <hip/hip_runtime.h>
#include <cmath>

typedef __attribute__((ext_vector_type(16))) _Float16 v16h;
typedef __attribute__((ext_vector_type(8)))  float    v8f;

__device__ __forceinline__ float elu1(float v) {
    return v > 0.0f ? v : expm1f(v);
}

__device__ __forceinline__ void atomAddF(float* p, float v) {
    __hip_atomic_fetch_add(p, v, __ATOMIC_RELAXED, __HIP_MEMORY_SCOPE_AGENT);
}

// ---------------- degree / normalization ----------------
__global__ void k_init_deg(float* deg, int N) {
    int i = blockIdx.x * blockDim.x + threadIdx.x;
    if (i < N) deg[i] = 1.0f;                       // self-loop contribution
}
__global__ void k_degree(const int* __restrict__ col, float* deg, int E) {
    int e = blockIdx.x * blockDim.x + threadIdx.x;
    if (e < E) atomAddF(&deg[col[e]], 1.0f);
}
__global__ void k_dinv(float* deg, int N) {
    int i = blockIdx.x * blockDim.x + threadIdx.x;
    if (i < N) deg[i] = rsqrtf(deg[i]);             // deg >= 1 always
}

// ---------------- WMMA GEMM: dst[N,64] = act(src + bias) @ W[K,64] ----------------
// Each wave owns a 16x32 output tile (2 column tiles) to keep B fragments +
// accumulators register-resident (no scratch spills). Waves pair by parity:
// even waves -> cols 0..31, odd waves -> cols 32..63, so hoisted B fragments
// are loop-invariant. Previous layer's bias+ELU fused into the A load.
template <int KC>   // K = KC*32 (KC=4 -> K=128, KC=2 -> K=64)
__global__ void __launch_bounds__(256, 1)
k_gemm(const float* __restrict__ src, const float* __restrict__ bias,
       const float* __restrict__ W, float* __restrict__ dst,
       int N, int applyAct) {
    constexpr int K = KC * 32;
    __shared__ _Float16 ldsW[K * 64];

    const int tid = threadIdx.x;
    for (int idx = tid; idx < K * 64; idx += blockDim.x)
        ldsW[idx] = (_Float16)W[idx];
    __syncthreads();

    const int lane = tid & 31;
    const int n    = lane & 15;      // col within tile (B/C/D) and row within tile (A)
    const int hi   = lane >> 4;

    const int nWaves = (gridDim.x * blockDim.x) >> 5;   // always even (256/block)
    const int gwave  = (blockIdx.x * blockDim.x + tid) >> 5;
    const int th     = gwave & 1;                       // column half: 0 -> 0..31, 1 -> 32..63

    // Hoist B fragments for this wave's 2 column tiles.
    // B treated as A-layout applied to B^T: lane holds column n, K striped per VGPR pair.
    v16h bfrag[2][KC];
    #pragma unroll
    for (int t = 0; t < 2; ++t)
        #pragma unroll
        for (int c = 0; c < KC; ++c)
            #pragma unroll
            for (int i = 0; i < 16; ++i) {
                int j  = i >> 1;
                int kl = ((j & 4) << 2) + (hi << 3) + ((j & 3) << 1) + (i & 1);
                bfrag[t][c][i] = ldsW[(c * 32 + kl) * 64 + (th * 2 + t) * 16 + n];
            }

    const int tiles  = N >> 4;                          // N % 16 == 0
    const int nPairs = nWaves >> 1;
    const int pair   = gwave >> 1;

    for (int tile = pair; tile < tiles; tile += nPairs) {
        const int rowA = tile * 16 + n;                 // A row for this lane
        const float* srow = src + (size_t)rowA * K;

        v8f acc[2] = {};
        #pragma unroll
        for (int c = 0; c < KC; ++c) {
            v16h a;
            #pragma unroll
            for (int j = 0; j < 8; ++j) {
                int kb = c * 32 + ((j & 4) << 2) + (hi << 3) + ((j & 3) << 1);
                float2 v = *(const float2*)(srow + kb);          // 8B aligned
                if (bias)     { v.x += bias[kb]; v.y += bias[kb + 1]; }
                if (applyAct) { v.x = elu1(v.x); v.y = elu1(v.y); }
                a[2 * j]     = (_Float16)v.x;
                a[2 * j + 1] = (_Float16)v.y;
            }
            #pragma unroll
            for (int t = 0; t < 2; ++t)
                acc[t] = __builtin_amdgcn_wmma_f32_16x16x32_f16(
                    false, a, false, bfrag[t][c], (short)0, acc[t], false, false);
        }

        // C/D layout: VGPR r -> row r (lanes 0-15) / r+8 (lanes 16-31), col = n
        #pragma unroll
        for (int t = 0; t < 2; ++t)
            #pragma unroll
            for (int r = 0; r < 8; ++r) {
                int rowg = tile * 16 + r + hi * 8;
                dst[(size_t)rowg * 64 + (th * 2 + t) * 16 + n] = acc[t][r];
            }
    }
}

// ---------------- edge scatter: out[col] += dinv[row]*dinv[col] * h[row] ----------------
// 16 threads per edge, float4 per thread. Edge ids >= E are self loops.
__global__ void k_aggregate(const float* __restrict__ h, const int* __restrict__ row,
                            const int* __restrict__ col, const float* __restrict__ dinv,
                            float* __restrict__ out, int E, int N) {
    long long gid   = (long long)blockIdx.x * blockDim.x + threadIdx.x;
    long long total = (long long)(E + N) * 16;
    if (gid >= total) return;
    int e = (int)(gid >> 4);
    int q = (int)(gid & 15);
    int r, c; float s;
    if (e < E) { r = row[e]; c = col[e]; s = dinv[r] * dinv[c]; }
    else       { r = c = e - E; float d = dinv[r]; s = d * d; }
    const float4 v = *(const float4*)(h + (size_t)r * 64 + q * 4);
    float* o = out + (size_t)c * 64 + q * 4;
    atomAddF(o + 0, s * v.x);
    atomAddF(o + 1, s * v.y);
    atomAddF(o + 2, s * v.z);
    atomAddF(o + 3, s * v.w);
}

// ---------------- readout: y = elu(agg + b3) @ Wl + bl ; segment sums/counts ----------------
__global__ void k_readout(const float* __restrict__ agg, const float* __restrict__ b3,
                          const float* __restrict__ Wl, const float* __restrict__ bl,
                          const int* __restrict__ batch, float* sums, float* cnts, int N) {
    int i = blockIdx.x * blockDim.x + threadIdx.x;
    if (i >= N) return;
    const float* hrow = agg + (size_t)i * 64;
    float acc = 0.0f;
    #pragma unroll 8
    for (int d = 0; d < 64; ++d)
        acc += elu1(hrow[d] + b3[d]) * Wl[d];
    float y = acc + bl[0];
    int g = batch[i];
    atomAddF(&sums[g], y);
    atomAddF(&cnts[g], 1.0f);
}

__global__ void k_pool(const float* __restrict__ sums, const float* __restrict__ cnts,
                       float* __restrict__ out, int G) {
    int g = blockIdx.x * blockDim.x + threadIdx.x;
    if (g < G) out[g] = sums[g] / fmaxf(cnts[g], 1.0f);
}

extern "C" void kernel_launch(void* const* d_in, const int* in_sizes, int n_in,
                              void* d_out, int out_size, void* d_ws, size_t ws_size,
                              hipStream_t stream) {
    const float* x     = (const float*)d_in[0];
    const int*   ei    = (const int*)d_in[1];
    const int*   batch = (const int*)d_in[2];
    const float* W1    = (const float*)d_in[3];
    const float* b1    = (const float*)d_in[4];
    const float* W2    = (const float*)d_in[5];
    const float* b2    = (const float*)d_in[6];
    const float* W3    = (const float*)d_in[7];
    const float* b3    = (const float*)d_in[8];
    const float* Wl    = (const float*)d_in[9];
    const float* bl    = (const float*)d_in[10];
    float* out = (float*)d_out;

    const int N   = in_sizes[2];
    const int E   = in_sizes[1] / 2;
    const int DIN = in_sizes[0] / N;     // 128
    const int G   = out_size;            // 64 graphs

    const int* row = ei;                 // edge_index[0] = sources
    const int* col = ei + E;             // edge_index[1] = targets

    float* ws     = (float*)d_ws;
    float* dinv   = ws;                              // N
    float* hbuf   = dinv + N;                        // N*64
    float* aggbuf = hbuf + (size_t)N * 64;           // N*64
    float* sums   = aggbuf + (size_t)N * 64;         // G sums + G counts
    (void)ws_size; (void)n_in;

    // --- degree / dinv (deg initialized to 1 for self loops) ---
    k_init_deg<<<(N + 255) / 256, 256, 0, stream>>>(dinv, N);
    hipMemsetAsync(sums, 0, sizeof(float) * 2 * G, stream);      // sums + cnts
    k_degree<<<(E + 255) / 256, 256, 0, stream>>>(col, dinv, E);
    k_dinv<<<(N + 255) / 256, 256, 0, stream>>>(dinv, N);

    const int GB = 128, GT = 256;        // 1024 waves = 512 pairs; ~6 row tiles each
    long long aggThreads = (long long)(E + N) * 16;
    int aggBlocks = (int)((aggThreads + 255) / 256);
    size_t featBytes = sizeof(float) * (size_t)N * 64;

    // --- layer 1: x @ W1 -> aggregate ---
    if (DIN == 128)
        k_gemm<4><<<GB, GT, 0, stream>>>(x, nullptr, W1, hbuf, N, 0);
    else
        k_gemm<2><<<GB, GT, 0, stream>>>(x, nullptr, W1, hbuf, N, 0);
    hipMemsetAsync(aggbuf, 0, featBytes, stream);
    k_aggregate<<<aggBlocks, 256, 0, stream>>>(hbuf, row, col, dinv, aggbuf, E, N);

    // --- layer 2: elu(agg1 + b1) @ W2 -> aggregate ---
    k_gemm<2><<<GB, GT, 0, stream>>>(aggbuf, b1, W2, hbuf, N, 1);
    hipMemsetAsync(aggbuf, 0, featBytes, stream);
    k_aggregate<<<aggBlocks, 256, 0, stream>>>(hbuf, row, col, dinv, aggbuf, E, N);

    // --- layer 3: elu(agg2 + b2) @ W3 -> aggregate ---
    k_gemm<2><<<GB, GT, 0, stream>>>(aggbuf, b2, W3, hbuf, N, 1);
    hipMemsetAsync(aggbuf, 0, featBytes, stream);
    k_aggregate<<<aggBlocks, 256, 0, stream>>>(hbuf, row, col, dinv, aggbuf, E, N);

    // --- readout + mean pool ---
    k_readout<<<(N + 255) / 256, 256, 0, stream>>>(aggbuf, b3, Wl, bl, batch,
                                                   sums, sums + G, N);
    k_pool<<<1, 256, 0, stream>>>(sums, sums + G, out, G);
}
